// ProductionMoE_68307159875976
// MI455X (gfx1250) — matrix operand order repaired and checked by reading, not compiled, not run on hardware
//
#include <hip/hip_runtime.h>
#include <hip/hip_bf16.h>

#define NTOK 8192
#define DIM  3
#define HDIM 1024
#define NEXP 8
#define KT   (HDIM/32)          // 32 k-tiles of 32
#define NT   (HDIM/16)          // 64 n-tiles of 16
#define FRAGS (NEXP*KT*NT)      // 16384 fragments per weight tensor

typedef __bf16 bf16;
typedef __attribute__((ext_vector_type(16))) __bf16 v16bf;
typedef __attribute__((ext_vector_type(8)))  __bf16 bf16x8;
typedef __attribute__((ext_vector_type(8)))  float  v8f;

#if defined(__AMDGCN__) && __has_builtin(__builtin_amdgcn_sched_group_barrier)
#define SGB(mask, size, id) __builtin_amdgcn_sched_group_barrier((mask), (size), (id))
#else
#define SGB(mask, size, id)
#endif

__global__ void zero_cnt_kernel(int* __restrict__ cnt) {
  if (threadIdx.x < NEXP) cnt[threadIdx.x] = 0;
}

// Convert w1/w2 (fp32 [E,H,H], contract-major) into bf16 WMMA-B fragment layout:
// fragment f = ((e*KT + kt)*NT + nt); per lane: 16 bf16 (K = 8*(lane>>4)+0..7 then 16+8*(lane>>4)+0..7, col = nt*16 + (lane&15))
// stored contiguously at f*512 + lane*16 so the GEMM does two global_load_b128 per lane.
__global__ __launch_bounds__(256) void pack_w_kernel(const float* __restrict__ w1,
                                                     const float* __restrict__ w2,
                                                     bf16* __restrict__ w1p,
                                                     bf16* __restrict__ w2p) {
  int gid  = blockIdx.x * 256 + threadIdx.x;
  int lane = gid & 31;
  int f    = gid >> 5;
  const float* src;
  bf16* dst;
  if (f >= FRAGS) { f -= FRAGS; src = w2; dst = w2p; }
  else            { src = w1;  dst = w1p; }
  int nt = f & (NT - 1);
  int kt = (f >> 6) & (KT - 1);
  int e  = f >> 11;
  int ncol  = nt * 16 + (lane & 15);
  int kbase = kt * 32 + 8 * (lane >> 4);
  const float* s = src + ((size_t)e * HDIM + kbase) * HDIM + ncol;
  bf16x8 lo, hi;
#pragma unroll
  for (int i = 0; i < 8; i++) lo[i] = (bf16)s[(size_t)i * HDIM];
#pragma unroll
  for (int i = 0; i < 8; i++) hi[i] = (bf16)s[(size_t)(i + 16) * HDIM];
  bf16* d = dst + (size_t)f * 512 + lane * 16;
  *(bf16x8*)d       = lo;
  *(bf16x8*)(d + 8) = hi;
}

// One block per token: h = relu(x@enc_w+enc_b), write bf16 h row, zero moe row,
// gate logits -> softmax -> top-2 -> normalized weights -> append to expert lists.
__global__ __launch_bounds__(256) void encode_route_kernel(
    const float* __restrict__ x, const float* __restrict__ encw, const float* __restrict__ encb,
    const float* __restrict__ gatew, const float* __restrict__ gateb,
    bf16* __restrict__ hbf, float* __restrict__ moe,
    int* __restrict__ cnt, int* __restrict__ listTok, float* __restrict__ listW) {
  const int n = blockIdx.x, t = threadIdx.x;
  __shared__ float hsh[HDIM];
  __shared__ float gred[NEXP * 256];
  const float x0 = x[n * DIM + 0], x1 = x[n * DIM + 1], x2 = x[n * DIM + 2];
#pragma unroll
  for (int j = t; j < HDIM; j += 256) {
    float v = encb[j];
    v = fmaf(x0, encw[0 * HDIM + j], v);
    v = fmaf(x1, encw[1 * HDIM + j], v);
    v = fmaf(x2, encw[2 * HDIM + j], v);
    v = fmaxf(v, 0.f);
    hsh[j] = v;
    hbf[(size_t)n * HDIM + j] = (bf16)v;
    moe[(size_t)n * HDIM + j] = 0.f;
  }
  __syncthreads();
  float p[NEXP];
#pragma unroll
  for (int e = 0; e < NEXP; e++) p[e] = 0.f;
  for (int j = t; j < HDIM; j += 256) {
    float hv = hsh[j];
#pragma unroll
    for (int e = 0; e < NEXP; e++) p[e] = fmaf(hv, gatew[j * NEXP + e], p[e]);
  }
#pragma unroll
  for (int e = 0; e < NEXP; e++) gred[e * 256 + t] = p[e];
  __syncthreads();
  for (int s = 128; s > 0; s >>= 1) {
    if (t < s) {
#pragma unroll
      for (int e = 0; e < NEXP; e++) gred[e * 256 + t] += gred[e * 256 + t + s];
    }
    __syncthreads();
  }
  if (t == 0) {
    float lg[NEXP], ex[NEXP];
    float m = -1e30f;
#pragma unroll
    for (int e = 0; e < NEXP; e++) { lg[e] = gred[e * 256] + gateb[e]; m = fmaxf(m, lg[e]); }
    float sum = 0.f;
#pragma unroll
    for (int e = 0; e < NEXP; e++) { ex[e] = __expf(lg[e] - m); sum += ex[e]; }
    int i0 = 0;
#pragma unroll
    for (int e = 1; e < NEXP; e++) if (ex[e] > ex[i0]) i0 = e;
    int i1 = (i0 == 0) ? 1 : 0;
#pragma unroll
    for (int e = 0; e < NEXP; e++) if (e != i0 && ex[e] > ex[i1]) i1 = e;
    float g0 = ex[i0], g1 = ex[i1];
    float inv = 1.f / (g0 + g1);
    g0 *= inv; g1 *= inv;
    int s0 = atomicAdd(&cnt[i0], 1);
    listTok[i0 * NTOK + s0] = n; listW[i0 * NTOK + s0] = g0;
    int s1 = atomicAdd(&cnt[i1], 1);
    listTok[i1 * NTOK + s1] = n; listW[i1 * NTOK + s1] = g1;
  }
}

// Routed expert FFN: block = (expert e, 16-token tile). 8 waves, each owns 128 output cols.
// Per k-step: batch all 18 loads, then 8 back-to-back WMMAs; sched_group_barrier pins
// the "loads-first, WMMAs-after" shape so waits become partial instead of loadcnt==0 stalls.
__global__ __launch_bounds__(256, 1) void moe_expert_kernel(
    const bf16* __restrict__ hbf, const bf16* __restrict__ w1p, const bf16* __restrict__ w2p,
    const float* __restrict__ b1, const float* __restrict__ b2,
    const int* __restrict__ cnt, const int* __restrict__ listTok, const float* __restrict__ listW,
    float* __restrict__ moe) {
  const int e    = blockIdx.x >> 9;     // 512 = NTOK/16 tiles per expert
  const int tile = blockIdx.x & 511;
  const int c    = cnt[e];
  if (tile * 16 >= c) return;

  __shared__ __align__(16) bf16 ehs[16 * HDIM];   // 32 KB intermediate tile
  __shared__ int   tokS[16];
  __shared__ float wS[16];

  const int t = threadIdx.x;
  if (t < 16) {
    int idx = tile * 16 + t;
    if (idx < c) { tokS[t] = listTok[e * NTOK + idx];      wS[t] = listW[e * NTOK + idx]; }
    else         { tokS[t] = listTok[e * NTOK + tile * 16]; wS[t] = 0.f; }
  }
  __syncthreads();

  const int lane = t & 31;
  const int wv   = t >> 5;      // wave id 0..7 -> cols [wv*128, wv*128+128)
  const int half = lane >> 4;
  const int l15  = lane & 15;
  const int tok  = tokS[l15];   // A-matrix row for this lane (M = lane&15)

  v8f acc[8];
#pragma unroll
  for (int j = 0; j < 8; j++) acc[j] = (v8f){0, 0, 0, 0, 0, 0, 0, 0};

  // ---- GEMM1: eh = relu(h @ w1[e] + b1[e]) ----
  {
    const bf16* aptr = hbf + (size_t)tok * HDIM + 8 * half;
    const bf16* bptr = w1p + ((size_t)(e * KT) * NT + wv * 8) * 512 + lane * 16;
    for (int kt = 0; kt < KT; ++kt) {
      // prefetch next k-step's B block toward the WGP (global_prefetch_b8)
      __builtin_prefetch(bptr + NT * 512, 0, 3);
      bf16x8 a0 = *(const bf16x8*)aptr;
      bf16x8 a1 = *(const bf16x8*)(aptr + 16);
      bf16x8 bb[16];
#pragma unroll
      for (int j = 0; j < 8; j++) {
        bb[2 * j]     = *(const bf16x8*)(bptr + j * 512);
        bb[2 * j + 1] = *(const bf16x8*)(bptr + j * 512 + 8);
      }
      v16bf A = __builtin_shufflevector(a0, a1, 0,1,2,3,4,5,6,7,8,9,10,11,12,13,14,15);
#pragma unroll
      for (int j = 0; j < 8; j++) {
        v16bf B = __builtin_shufflevector(bb[2 * j], bb[2 * j + 1],
                                          0,1,2,3,4,5,6,7,8,9,10,11,12,13,14,15);
        acc[j] = __builtin_amdgcn_wmma_f32_16x16x32_bf16(false, A, false, B, (short)0,
                                                         acc[j], false, false);
      }
      // pin schedule: issue this k-step's 18 VMEM reads first, then the 8 WMMAs
      SGB(0x020, 18, 0);   // VMEM read group
      SGB(0x008, 8, 0);    // MFMA/WMMA group
      aptr += 32;
      bptr += NT * 512;
    }
  }
  // epilogue 1: bias + relu, stage as bf16 in LDS (D layout: row M = r + 8*half, col = nt*16 + l15)
#pragma unroll
  for (int j = 0; j < 8; j++) {
    int col = (wv * 8 + j) * 16 + l15;
    float bias = b1[e * HDIM + col];
#pragma unroll
    for (int r = 0; r < 8; r++) {
      int m = r + 8 * half;
      float v = fmaxf(acc[j][r] + bias, 0.f);
      ehs[m * HDIM + col] = (bf16)v;
    }
  }
  __syncthreads();

#pragma unroll
  for (int j = 0; j < 8; j++) acc[j] = (v8f){0, 0, 0, 0, 0, 0, 0, 0};

  // ---- GEMM2: eo = eh @ w2[e] + b2[e] ----
  {
    const bf16* aptr = &ehs[l15 * HDIM + 8 * half];
    const bf16* bptr = w2p + ((size_t)(e * KT) * NT + wv * 8) * 512 + lane * 16;
    for (int kt = 0; kt < KT; ++kt) {
      __builtin_prefetch(bptr + NT * 512, 0, 3);
      bf16x8 a0 = *(const bf16x8*)aptr;          // ds_load_b128
      bf16x8 a1 = *(const bf16x8*)(aptr + 16);
      bf16x8 bb[16];
#pragma unroll
      for (int j = 0; j < 8; j++) {
        bb[2 * j]     = *(const bf16x8*)(bptr + j * 512);
        bb[2 * j + 1] = *(const bf16x8*)(bptr + j * 512 + 8);
      }
      v16bf A = __builtin_shufflevector(a0, a1, 0,1,2,3,4,5,6,7,8,9,10,11,12,13,14,15);
#pragma unroll
      for (int j = 0; j < 8; j++) {
        v16bf B = __builtin_shufflevector(bb[2 * j], bb[2 * j + 1],
                                          0,1,2,3,4,5,6,7,8,9,10,11,12,13,14,15);
        acc[j] = __builtin_amdgcn_wmma_f32_16x16x32_bf16(false, A, false, B, (short)0,
                                                         acc[j], false, false);
      }
      // pin schedule: 2 DS reads + 16 VMEM reads first, then the 8 WMMAs
      SGB(0x100, 2, 0);    // DS read group
      SGB(0x020, 16, 0);   // VMEM read group
      SGB(0x008, 8, 0);    // MFMA/WMMA group
      aptr += 32;
      bptr += NT * 512;
    }
  }
  // epilogue 2: bias, scale by combine weight, accumulate into moe (exactly 2 contributions/element)
#pragma unroll
  for (int j = 0; j < 8; j++) {
    int col = (wv * 8 + j) * 16 + l15;
    float bias = b2[e * HDIM + col];
#pragma unroll
    for (int r = 0; r < 8; r++) {
      int m = r + 8 * half;
      float w = wS[m];
      if (w != 0.f) {
        float v = (acc[j][r] + bias) * w;
        atomicAdd(&moe[(size_t)tokS[m] * HDIM + col], v);
      }
    }
  }
}

// out = moe @ dec_w + dec_b  ([N,1024] x [1024,3])
__global__ __launch_bounds__(128) void decode_kernel(const float* __restrict__ moe,
                                                     const float* __restrict__ decw,
                                                     const float* __restrict__ decb,
                                                     float* __restrict__ out) {
  const int n = blockIdx.x, t = threadIdx.x;
  float s0 = 0.f, s1 = 0.f, s2 = 0.f;
  for (int j = t; j < HDIM; j += 128) {
    float m = moe[(size_t)n * HDIM + j];
    s0 = fmaf(m, decw[j * DIM + 0], s0);
    s1 = fmaf(m, decw[j * DIM + 1], s1);
    s2 = fmaf(m, decw[j * DIM + 2], s2);
  }
  __shared__ float red[3][128];
  red[0][t] = s0; red[1][t] = s1; red[2][t] = s2;
  __syncthreads();
  for (int s = 64; s > 0; s >>= 1) {
    if (t < s) {
      red[0][t] += red[0][t + s];
      red[1][t] += red[1][t + s];
      red[2][t] += red[2][t + s];
    }
    __syncthreads();
  }
  if (t == 0) {
    out[n * DIM + 0] = red[0][0] + decb[0];
    out[n * DIM + 1] = red[1][0] + decb[1];
    out[n * DIM + 2] = red[2][0] + decb[2];
  }
}

extern "C" void kernel_launch(void* const* d_in, const int* in_sizes, int n_in,
                              void* d_out, int out_size, void* d_ws, size_t ws_size,
                              hipStream_t stream) {
  (void)in_sizes; (void)n_in; (void)out_size; (void)ws_size;
  const float* x     = (const float*)d_in[0];
  const float* encw  = (const float*)d_in[1];
  const float* encb  = (const float*)d_in[2];
  const float* w1    = (const float*)d_in[3];
  const float* b1    = (const float*)d_in[4];
  const float* w2    = (const float*)d_in[5];
  const float* b2    = (const float*)d_in[6];
  const float* gatew = (const float*)d_in[7];
  const float* gateb = (const float*)d_in[8];
  const float* decw  = (const float*)d_in[9];
  const float* decb  = (const float*)d_in[10];
  float* out = (float*)d_out;

  char* ws = (char*)d_ws;
  size_t off = 0;
  auto take = [&](size_t bytes) -> char* {
    char* p = ws + off;
    off = (off + bytes + 255) & ~(size_t)255;
    return p;
  };
  bf16*  hbf     = (bf16*)take((size_t)NTOK * HDIM * sizeof(bf16));           // 16.8 MB
  bf16*  w1p     = (bf16*)take((size_t)NEXP * HDIM * HDIM * sizeof(bf16));    // 16.8 MB
  bf16*  w2p     = (bf16*)take((size_t)NEXP * HDIM * HDIM * sizeof(bf16));    // 16.8 MB
  float* moe     = (float*)take((size_t)NTOK * HDIM * sizeof(float));         // 33.6 MB
  int*   cnt     = (int*)take(NEXP * sizeof(int));
  int*   listTok = (int*)take((size_t)NEXP * NTOK * sizeof(int));
  float* listW   = (float*)take((size_t)NEXP * NTOK * sizeof(float));

  zero_cnt_kernel<<<1, 32, 0, stream>>>(cnt);
  pack_w_kernel<<<(2 * FRAGS * 32) / 256, 256, 0, stream>>>(w1, w2, w1p, w2p);
  encode_route_kernel<<<NTOK, 256, 0, stream>>>(x, encw, encb, gatew, gateb,
                                                hbf, moe, cnt, listTok, listW);
  moe_expert_kernel<<<NEXP * (NTOK / 16), 256, 0, stream>>>(hbf, w1p, w2p, b1, b2,
                                                            cnt, listTok, listW, moe);
  decode_kernel<<<NTOK, 128, 0, stream>>>(moe, decw, decb, out);
}
